// VisualQuestionEncoder_12360915878450
// MI455X (gfx1250) — compile-verified, matrix operand order checked by reading, not run on hardware
//
#include <hip/hip_runtime.h>
#include <hip/hip_bf16.h>

// ---------------- problem constants ----------------
#define BS   128
#define NP   256
#define NR   10
#define DIM  1024
#define CE   512
#define TE   512
#define NH   8
#define HD   128
#define TOPK 16
#define ROWS_Q  (BS * NR)   // 1280
#define ROWS_KV (BS * NP)   // 32768

typedef unsigned short ushort_t;
typedef unsigned char  uchar_t;
typedef unsigned long long ull_t;

typedef __bf16 bf16x16 __attribute__((ext_vector_type(16)));
typedef __bf16 bf16x8  __attribute__((ext_vector_type(8)));
typedef float  f32x8   __attribute__((ext_vector_type(8)));

union BV { bf16x16 v; bf16x8 h[2]; };

// ---------------- small helpers ----------------
static __device__ __forceinline__ ushort_t f2bf(float f) {
    union { float f; unsigned u; } v; v.f = f;
    unsigned r = v.u + 0x7FFFu + ((v.u >> 16) & 1u);   // round-to-nearest-even
    return (ushort_t)(r >> 16);
}
static __device__ __forceinline__ float bf2f(ushort_t h) {
    union { unsigned u; float f; } v; v.u = ((unsigned)h) << 16; return v.f;
}
static __device__ __forceinline__ float bflo(unsigned u) {
    union { unsigned x; float f; } v; v.x = u << 16; return v.f;
}
static __device__ __forceinline__ float bfhi(unsigned u) {
    union { unsigned x; float f; } v; v.x = u & 0xFFFF0000u; return v.f;
}
static __device__ __forceinline__ float dot8(uint4 a, uint4 b) {
    float s;
    s  = bflo(a.x) * bflo(b.x) + bfhi(a.x) * bfhi(b.x);
    s += bflo(a.y) * bflo(b.y) + bfhi(a.y) * bfhi(b.y);
    s += bflo(a.z) * bflo(b.z) + bfhi(a.z) * bfhi(b.z);
    s += bflo(a.w) * bflo(b.w) + bfhi(a.w) * bfhi(b.w);
    return s;
}
static __device__ __forceinline__ float block_reduce_sum(float v, float* red) {
    int t = threadIdx.x;
    red[t] = v; __syncthreads();
    for (int s = 128; s > 0; s >>= 1) { if (t < s) red[t] += red[t + s]; __syncthreads(); }
    float r = red[0]; __syncthreads();
    return r;
}
static __device__ __forceinline__ float block_reduce_max(float v, float* red) {
    int t = threadIdx.x;
    red[t] = v; __syncthreads();
    for (int s = 128; s > 0; s >>= 1) { if (t < s) red[t] = fmaxf(red[t], red[t + s]); __syncthreads(); }
    float r = red[0]; __syncthreads();
    return r;
}

// ---------------- prep kernels ----------------
__global__ __launch_bounds__(256) void cvt_f32_bf16_kernel(const float* __restrict__ in,
                                                           ushort_t* __restrict__ out, int n) {
    int i = blockIdx.x * blockDim.x + threadIdx.x;
    int stride = gridDim.x * blockDim.x;
    for (; i < n; i += stride) out[i] = f2bf(in[i]);
}

// LN over rows of 1024 f32 -> bf16 (also used for img_emb directly)
__global__ __launch_bounds__(256) void ln_kernel(const float* __restrict__ x,
                                                 const float* __restrict__ g,
                                                 const float* __restrict__ b,
                                                 ushort_t* __restrict__ out) {
    __shared__ float red[256];
    int row = blockIdx.x, t = threadIdx.x;
    const float4* xr = (const float4*)(x + (size_t)row * DIM);
    float4 v = xr[t];
    float mean = block_reduce_sum(v.x + v.y + v.z + v.w, red) * (1.0f / DIM);
    float d0 = v.x - mean, d1 = v.y - mean, d2 = v.z - mean, d3 = v.w - mean;
    float var = block_reduce_sum(d0 * d0 + d1 * d1 + d2 * d2 + d3 * d3, red) * (1.0f / DIM);
    float rstd = rsqrtf(var + 1e-5f);
    int c = t * 4;
    size_t o = (size_t)row * DIM + c;
    out[o + 0] = f2bf(d0 * rstd * g[c + 0] + b[c + 0]);
    out[o + 1] = f2bf(d1 * rstd * g[c + 1] + b[c + 1]);
    out[o + 2] = f2bf(d2 * rstd * g[c + 2] + b[c + 2]);
    out[o + 3] = f2bf(d3 * rstd * g[c + 3] + b[c + 3]);
}

// build contA bf16 [1280,512], catA bf16 [1280,1024], and write cat_emb f32 into out[:,1024:2048]
__global__ __launch_bounds__(256) void prep_cat_kernel(const float* __restrict__ cont,
                                                       const float* __restrict__ type,
                                                       ushort_t* __restrict__ contA,
                                                       ushort_t* __restrict__ catA,
                                                       float* __restrict__ dout) {
    int row = blockIdx.x, t = threadIdx.x;
    for (int c = t; c < CE; c += 256) {
        float cv = cont[(size_t)row * CE + c];
        float tv = type[(size_t)row * TE + c];
        contA[(size_t)row * CE + c]       = f2bf(cv);
        catA[(size_t)row * DIM + c]       = f2bf(cv);
        catA[(size_t)row * DIM + CE + c]  = f2bf(tv);
        dout[(size_t)row * 2048 + 1024 + c]       = cv;
        dout[(size_t)row * 2048 + 1024 + CE + c]  = tv;
    }
}

// ---------------- WMMA NT-GEMM: C[M,N] = A[M,K](bf16) * Bw[N,K](bf16)^T + bias (+resid) ----------------
// block = 256 threads (8 waves); block tile 256(M) x 64(N); wave tile 32 x 64
// (2 A row-stripes x 4 N-tiles -> 8 accumulators; 12 b128 loads per 8 WMMAs); K step 32.
template<bool OUTF32, bool GELU_ACT, bool HASRES>
__global__ __launch_bounds__(256) void gemm_nt_kernel(const ushort_t* __restrict__ A,
                                                      const ushort_t* __restrict__ Bw,
                                                      const float* __restrict__ bias,
                                                      const float* __restrict__ resid,
                                                      float* __restrict__ Cf,
                                                      ushort_t* __restrict__ Cb,
                                                      int M, int N, int K, int ldc) {
    const int wave = threadIdx.x >> 5;
    const int lane = threadIdx.x & 31;
    const int l15  = lane & 15;
    const int klane = (lane >> 4) << 3;                 // 0 or 8
    const int mrow0 = blockIdx.y * 256 + wave * 32 + l15;   // A stripe 0
    const int nbase = blockIdx.x * 64;

    f32x8 zero = {0.f, 0.f, 0.f, 0.f, 0.f, 0.f, 0.f, 0.f};
    f32x8 acc[2][4] = {{zero, zero, zero, zero}, {zero, zero, zero, zero}};

    const ushort_t* arow0 = A + (size_t)mrow0 * K + klane;
    const ushort_t* arow1 = arow0 + (size_t)16 * K;          // A stripe 1 (rows +16)
    const ushort_t* b0 = Bw + (size_t)(nbase +  0 + l15) * K + klane;
    const ushort_t* b1 = Bw + (size_t)(nbase + 16 + l15) * K + klane;
    const ushort_t* b2 = Bw + (size_t)(nbase + 32 + l15) * K + klane;
    const ushort_t* b3 = Bw + (size_t)(nbase + 48 + l15) * K + klane;

    for (int k0 = 0; k0 < K; k0 += 32) {
        BV a0, a1;
        a0.h[0] = *(const bf16x8*)(arow0 + k0);
        a0.h[1] = *(const bf16x8*)(arow0 + k0 + 16);
        a1.h[0] = *(const bf16x8*)(arow1 + k0);
        a1.h[1] = *(const bf16x8*)(arow1 + k0 + 16);
        BV bb0, bb1, bb2, bb3;
        bb0.h[0] = *(const bf16x8*)(b0 + k0); bb0.h[1] = *(const bf16x8*)(b0 + k0 + 16);
        bb1.h[0] = *(const bf16x8*)(b1 + k0); bb1.h[1] = *(const bf16x8*)(b1 + k0 + 16);
        bb2.h[0] = *(const bf16x8*)(b2 + k0); bb2.h[1] = *(const bf16x8*)(b2 + k0 + 16);
        bb3.h[0] = *(const bf16x8*)(b3 + k0); bb3.h[1] = *(const bf16x8*)(b3 + k0 + 16);
        acc[0][0] = __builtin_amdgcn_wmma_f32_16x16x32_bf16(false, a0.v, false, bb0.v, (short)0, acc[0][0], false, false);
        acc[0][1] = __builtin_amdgcn_wmma_f32_16x16x32_bf16(false, a0.v, false, bb1.v, (short)0, acc[0][1], false, false);
        acc[0][2] = __builtin_amdgcn_wmma_f32_16x16x32_bf16(false, a0.v, false, bb2.v, (short)0, acc[0][2], false, false);
        acc[0][3] = __builtin_amdgcn_wmma_f32_16x16x32_bf16(false, a0.v, false, bb3.v, (short)0, acc[0][3], false, false);
        acc[1][0] = __builtin_amdgcn_wmma_f32_16x16x32_bf16(false, a1.v, false, bb0.v, (short)0, acc[1][0], false, false);
        acc[1][1] = __builtin_amdgcn_wmma_f32_16x16x32_bf16(false, a1.v, false, bb1.v, (short)0, acc[1][1], false, false);
        acc[1][2] = __builtin_amdgcn_wmma_f32_16x16x32_bf16(false, a1.v, false, bb2.v, (short)0, acc[1][2], false, false);
        acc[1][3] = __builtin_amdgcn_wmma_f32_16x16x32_bf16(false, a1.v, false, bb3.v, (short)0, acc[1][3], false, false);
    }

    // C layout: VGPR r, lanes 0-15 -> M=r, N=lane; lanes 16-31 -> M=8+r, N=lane-16
    const int rbase = blockIdx.y * 256 + wave * 32 + ((lane >> 4) << 3);
#pragma unroll
    for (int st = 0; st < 2; ++st) {
#pragma unroll
        for (int r = 0; r < 8; ++r) {
            int row = rbase + st * 16 + r;
#pragma unroll
            for (int j = 0; j < 4; ++j) {
                int col = nbase + j * 16 + l15;
                float v = acc[st][j][r] + bias[col];
                if (HASRES) v += resid[(size_t)row * N + col];
                if (GELU_ACT) {
                    float x3 = v * v * v;
                    v = 0.5f * v * (1.0f + tanhf(0.7978845608028654f * (v + 0.044715f * x3)));
                }
                if (OUTF32) Cf[(size_t)row * ldc + col] = v;
                else        Cb[(size_t)row * ldc + col] = f2bf(v);
            }
        }
    }
}

// ---------------- stage-1 attention weights + top-16 -> irrelevance mask ----------------
// grid = BS*NR blocks, 256 threads (thread t <-> proposal t)
__global__ __launch_bounds__(256) void attn1_topk_kernel(const ushort_t* __restrict__ qp,
                                                         const ushort_t* __restrict__ kp,
                                                         const uchar_t* __restrict__ pad,
                                                         uchar_t* __restrict__ irr) {
    __shared__ float red[256];
    __shared__ ull_t kred[256];
    int bq = blockIdx.x;
    int b  = bq / NR;
    int t  = threadIdx.x;

    const ushort_t* qrow = qp + (size_t)bq * DIM;
    const ushort_t* krow = kp + ((size_t)b * NP + t) * DIM;
    bool padded = pad[b * NP + t] != 0;

    float aw = 0.0f;
    for (int h = 0; h < NH; ++h) {
        const uint4* q4 = (const uint4*)(qrow + h * HD);
        const uint4* k4 = (const uint4*)(krow + h * HD);
        float s = 0.0f;
#pragma unroll
        for (int i = 0; i < 16; ++i) s += dot8(q4[i], k4[i]);
        s *= 0.08838834764831845f;                    // 1/sqrt(128)
        if (padded) s = -1e9f;
        float m  = block_reduce_max(s, red);
        float e  = __expf(s - m);
        float sm = block_reduce_sum(e, red);
        aw += e / sm;
    }
    aw *= (1.0f / NH);

    // iterative top-16 via packed argmax reductions
    float myval = aw;
    bool  mysel = false;
    for (int it = 0; it < TOPK; ++it) {
        unsigned ub = __float_as_uint(myval);
        ub = (ub & 0x80000000u) ? ~ub : (ub | 0x80000000u);     // order-preserving map
        kred[t] = (((ull_t)ub) << 32) | (unsigned)t;
        __syncthreads();
        for (int s = 128; s > 0; s >>= 1) {
            if (t < s) kred[t] = (kred[t] > kred[t + s]) ? kred[t] : kred[t + s];
            __syncthreads();
        }
        int widx = (int)(kred[0] & 0xFFFFFFFFull);
        if (t == widx) { mysel = true; myval = -1e30f; }
        __syncthreads();
    }
    irr[(size_t)bq * NP + t] = (padded || !mysel) ? 1 : 0;
}

// ---------------- stage-2 masked attention -> per-(b,r) context [DIM] bf16 ----------------
__global__ __launch_bounds__(256) void attn2_ctx_kernel(const ushort_t* __restrict__ qp,
                                                        const ushort_t* __restrict__ kp,
                                                        const ushort_t* __restrict__ vp,
                                                        const uchar_t* __restrict__ irr,
                                                        ushort_t* __restrict__ ctx) {
    __shared__ float red[256];
    __shared__ float ps[256];
    __shared__ float pf[256];
    int bq = blockIdx.x;
    int b  = bq / NR;
    int t  = threadIdx.x;

    const ushort_t* qrow = qp + (size_t)bq * DIM;
    const ushort_t* krow = kp + ((size_t)b * NP + t) * DIM;
    bool masked = irr[(size_t)bq * NP + t] != 0;

    for (int h = 0; h < NH; ++h) {
        const uint4* q4 = (const uint4*)(qrow + h * HD);
        const uint4* k4 = (const uint4*)(krow + h * HD);
        float s = 0.0f;
#pragma unroll
        for (int i = 0; i < 16; ++i) s += dot8(q4[i], k4[i]);
        s *= 0.08838834764831845f;
        if (masked) s = -1e9f;
        float m  = block_reduce_max(s, red);
        float e  = __expf(s - m);
        float sm = block_reduce_sum(e, red);
        ps[t] = e / sm;
        __syncthreads();

        // context: ctx[h*128+d] = sum_k p[k] * V[b,k,h*128+d]; split k-range over 2 half-blocks
        int d = t & 127, half = t >> 7;
        const ushort_t* vcol = vp + ((size_t)b * NP + half * 128) * DIM + h * HD + d;
        float acc = 0.0f;
#pragma unroll 4
        for (int k2 = 0; k2 < 128; ++k2)
            acc += ps[half * 128 + k2] * bf2f(vcol[(size_t)k2 * DIM]);
        pf[t] = acc;
        __syncthreads();
        if (t < 128)
            ctx[(size_t)bq * DIM + h * HD + t] = f2bf(pf[t] + pf[t + 128]);
        __syncthreads();
    }
}

// ---------------- host launcher ----------------
extern "C" void kernel_launch(void* const* d_in, const int* in_sizes, int n_in,
                              void* d_out, int out_size, void* d_ws, size_t ws_size,
                              hipStream_t stream) {
    (void)in_sizes; (void)n_in; (void)out_size; (void)ws_size;

    const float*   img_emb   = (const float*)d_in[0];
    const uchar_t* img_masks = (const uchar_t*)d_in[1];
    const float*   cont_emb  = (const float*)d_in[2];
    const float*   type_emb  = (const float*)d_in[3];
    const float*   ln_g      = (const float*)d_in[4];
    const float*   ln_b      = (const float*)d_in[5];
    const float*   wcont_w   = (const float*)d_in[6];
    const float*   wcont_b   = (const float*)d_in[7];
    const float*   wcat_w    = (const float*)d_in[8];
    const float*   wcat_b    = (const float*)d_in[9];
    const float*   cont_in_w = (const float*)d_in[10];
    const float*   cont_in_b = (const float*)d_in[11];
    // d_in[12], d_in[13]: cont_out_w/b are dead code in the reference (attn output discarded)
    const float*   cat_in_w  = (const float*)d_in[14];
    const float*   cat_in_b  = (const float*)d_in[15];
    const float*   cat_out_w = (const float*)d_in[16];
    const float*   cat_out_b = (const float*)d_in[17];
    const float*   pnn_w1    = (const float*)d_in[18];
    const float*   pnn_b1    = (const float*)d_in[19];
    const float*   pnn_w2    = (const float*)d_in[20];
    const float*   pnn_b2    = (const float*)d_in[21];
    float* out = (float*)d_out;

    // workspace carving (256B aligned)
    char* ws = (char*)d_ws;
    size_t off = 0;
    auto alloc = [&](size_t bytes) -> char* {
        char* p = ws + off;
        off = (off + bytes + 255) & ~(size_t)255;
        return p;
    };
    ushort_t* wcont_bf  = (ushort_t*)alloc((size_t)DIM * CE * 2);
    ushort_t* wcat_bf   = (ushort_t*)alloc((size_t)DIM * DIM * 2);
    ushort_t* wqk1_bf   = (ushort_t*)alloc((size_t)2 * DIM * DIM * 2);   // stage1 Wq,Wk
    ushort_t* wqkv2_bf  = (ushort_t*)alloc((size_t)3 * DIM * DIM * 2);   // stage2 Wq,Wk,Wv
    ushort_t* wout2_bf  = (ushort_t*)alloc((size_t)DIM * DIM * 2);
    ushort_t* pw1_bf    = (ushort_t*)alloc((size_t)DIM * DIM * 2);
    ushort_t* pw2_bf    = (ushort_t*)alloc((size_t)DIM * DIM * 2);
    ushort_t* contA     = (ushort_t*)alloc((size_t)ROWS_Q * CE * 2);
    ushort_t* catA      = (ushort_t*)alloc((size_t)ROWS_Q * DIM * 2);
    float*    cont_q    = (float*)   alloc((size_t)ROWS_Q * DIM * 4);
    float*    cat_q     = (float*)   alloc((size_t)ROWS_Q * DIM * 4);
    ushort_t* cont_nbf  = (ushort_t*)alloc((size_t)ROWS_Q * DIM * 2);
    ushort_t* cat_nbf   = (ushort_t*)alloc((size_t)ROWS_Q * DIM * 2);
    ushort_t* qp1       = (ushort_t*)alloc((size_t)ROWS_Q * DIM * 2);
    ushort_t* qp2       = (ushort_t*)alloc((size_t)ROWS_Q * DIM * 2);
    uchar_t*  irr       = (uchar_t*) alloc((size_t)ROWS_Q * NP);
    ushort_t* ctx       = (ushort_t*)alloc((size_t)ROWS_Q * DIM * 2);
    float*    xres      = (float*)   alloc((size_t)ROWS_Q * DIM * 4);
    ushort_t* xn_bf     = (ushort_t*)alloc((size_t)ROWS_Q * DIM * 2);
    ushort_t* h1_bf     = (ushort_t*)alloc((size_t)ROWS_Q * DIM * 2);
    ushort_t* img_nbf   = (ushort_t*)alloc((size_t)ROWS_KV * DIM * 2);
    ushort_t* kp_bf     = (ushort_t*)alloc((size_t)ROWS_KV * DIM * 2);  // stage1 K, reused for stage2 K
    ushort_t* vp_bf     = (ushort_t*)alloc((size_t)ROWS_KV * DIM * 2);

    dim3 blk(256);
    auto cvt = [&](const float* src, ushort_t* dst, int n) {
        cvt_f32_bf16_kernel<<<dim3((n + 255) / 256), blk, 0, stream>>>(src, dst, n);
    };

    // 1) weight conversions to bf16
    cvt(wcont_w,   wcont_bf,  DIM * CE);
    cvt(wcat_w,    wcat_bf,   DIM * DIM);
    cvt(cont_in_w, wqk1_bf,   2 * DIM * DIM);     // only Wq,Wk needed for stage1
    cvt(cat_in_w,  wqkv2_bf,  3 * DIM * DIM);
    cvt(cat_out_w, wout2_bf,  DIM * DIM);
    cvt(pnn_w1,    pw1_bf,    DIM * DIM);
    cvt(pnn_w2,    pw2_bf,    DIM * DIM);

    // 2) img_norm = LN(img_emb) -> bf16 ; prep cont/cat activations + output tail (cat_emb)
    ln_kernel<<<dim3(ROWS_KV), blk, 0, stream>>>(img_emb, ln_g, ln_b, img_nbf);
    prep_cat_kernel<<<dim3(ROWS_Q), blk, 0, stream>>>(cont_emb, type_emb, contA, catA, out);

    dim3 gQ(DIM / 64, ROWS_Q / 256);    // (16, 5)
    dim3 gKV(DIM / 64, ROWS_KV / 256);  // (16, 128)

    // 3) cont_q / cat_q (f32 for later LN & residual)
    gemm_nt_kernel<true, false, false><<<gQ, blk, 0, stream>>>(contA, wcont_bf, wcont_b, nullptr,
                                                               cont_q, nullptr, ROWS_Q, DIM, CE, DIM);
    gemm_nt_kernel<true, false, false><<<gQ, blk, 0, stream>>>(catA, wcat_bf, wcat_b, nullptr,
                                                               cat_q, nullptr, ROWS_Q, DIM, DIM, DIM);
    ln_kernel<<<dim3(ROWS_Q), blk, 0, stream>>>(cont_q, ln_g, ln_b, cont_nbf);
    ln_kernel<<<dim3(ROWS_Q), blk, 0, stream>>>(cat_q, ln_g, ln_b, cat_nbf);

    // 4) stage-1 projections (Q over 1280 rows, K over 32768 rows; V/out-proj are dead code)
    gemm_nt_kernel<false, false, false><<<gQ, blk, 0, stream>>>(cont_nbf, wqk1_bf, cont_in_b, nullptr,
                                                                nullptr, qp1, ROWS_Q, DIM, DIM, DIM);
    gemm_nt_kernel<false, false, false><<<gKV, blk, 0, stream>>>(img_nbf, wqk1_bf + (size_t)DIM * DIM,
                                                                 cont_in_b + DIM, nullptr,
                                                                 nullptr, kp_bf, ROWS_KV, DIM, DIM, DIM);
    // 5) head-averaged attention + top-16 -> irrelevance mask
    attn1_topk_kernel<<<dim3(ROWS_Q), blk, 0, stream>>>(qp1, kp_bf, img_masks, irr);

    // 6) stage-2 projections (kp_bf reused; stream order guarantees stage-1 consumer ran first)
    gemm_nt_kernel<false, false, false><<<gQ, blk, 0, stream>>>(cat_nbf, wqkv2_bf, cat_in_b, nullptr,
                                                                nullptr, qp2, ROWS_Q, DIM, DIM, DIM);
    gemm_nt_kernel<false, false, false><<<gKV, blk, 0, stream>>>(img_nbf, wqkv2_bf + (size_t)DIM * DIM,
                                                                 cat_in_b + DIM, nullptr,
                                                                 nullptr, kp_bf, ROWS_KV, DIM, DIM, DIM);
    gemm_nt_kernel<false, false, false><<<gKV, blk, 0, stream>>>(img_nbf, wqkv2_bf + (size_t)2 * DIM * DIM,
                                                                 cat_in_b + 2 * DIM, nullptr,
                                                                 nullptr, vp_bf, ROWS_KV, DIM, DIM, DIM);
    // 7) masked attention -> context
    attn2_ctx_kernel<<<dim3(ROWS_Q), blk, 0, stream>>>(qp2, kp_bf, vp_bf, irr, ctx);

    // 8) out-proj + residual cat_q -> x (f32); LN; GELU-FFN; final write into d_out (ldc = 2048)
    gemm_nt_kernel<true, false, true><<<gQ, blk, 0, stream>>>(ctx, wout2_bf, cat_out_b, cat_q,
                                                              xres, nullptr, ROWS_Q, DIM, DIM, DIM);
    ln_kernel<<<dim3(ROWS_Q), blk, 0, stream>>>(xres, ln_g, ln_b, xn_bf);
    gemm_nt_kernel<false, true, false><<<gQ, blk, 0, stream>>>(xn_bf, pw1_bf, pnn_b1, nullptr,
                                                               nullptr, h1_bf, ROWS_Q, DIM, DIM, DIM);
    gemm_nt_kernel<true, false, true><<<gQ, blk, 0, stream>>>(h1_bf, pw2_bf, pnn_b2, xres,
                                                              out, nullptr, ROWS_Q, DIM, DIM, 2048);
}
